// VS_DiT_Block_52072183497519
// MI455X (gfx1250) — matrix operand order repaired and checked by reading, not compiled, not run on hardware
//
#include <hip/hip_runtime.h>
#include <cmath>

// ---------------------------------------------------------------------------
// Types for CDNA5 WMMA (wave32): V_WMMA_F32_16X16X32_BF16
// ---------------------------------------------------------------------------
typedef __attribute__((ext_vector_type(16))) __bf16 v16bf;
typedef __attribute__((ext_vector_type(8)))  float  v8f;

union BF16Frag {
    v16bf v;
    uint4 q[2];
    unsigned short h[16];
};

__device__ __forceinline__ v8f wmma_bf16(const BF16Frag& a, const BF16Frag& b, v8f c) {
    return __builtin_amdgcn_wmma_f32_16x16x32_bf16(false, a.v, false, b.v,
                                                   (short)0, c, false, false);
}

__device__ __forceinline__ unsigned short f2bf(float f) {
    unsigned int u = __float_as_uint(f);
    unsigned int r = u + 0x7FFFu + ((u >> 16) & 1u);   // round-to-nearest-even
    return (unsigned short)(r >> 16);
}

// ---------------------------------------------------------------------------
// Async global->LDS via inline asm (GLOBAL_LOAD_ASYNC_TO_LDS_B128, ASYNCcnt).
// VGLOBAL encoding: VDST = LDS byte address, VADDR = 64-bit global address (GV).
// Generic->LDS address: low 32 bits of the flat pointer are the LDS offset.
// ---------------------------------------------------------------------------
#ifndef USE_ASYNC_LDS
#define USE_ASYNC_LDS 1
#endif

#if USE_ASYNC_LDS
__device__ __forceinline__ void async_g2l_b128(const void* g, void* l) {
    unsigned long long ga = (unsigned long long)g;
    unsigned int la = (unsigned int)(unsigned long long)l;
    asm volatile("global_load_async_to_lds_b128 %0, %1, off"
                 :: "v"(la), "v"(ga) : "memory");
}
__device__ __forceinline__ void wait_async0() {
    asm volatile("s_wait_asynccnt 0" ::: "memory");
}
#endif

// ---------------------------------------------------------------------------
// fp32 -> bf16 conversion (weights, context)
// ---------------------------------------------------------------------------
__global__ __launch_bounds__(256)
void f32_to_bf16_kernel(const float* __restrict__ in, unsigned short* __restrict__ out, long n) {
    long i = (long)blockIdx.x * blockDim.x + threadIdx.x;
    long stride = (long)gridDim.x * blockDim.x;
    for (; i < n; i += stride) out[i] = f2bf(in[i]);
}

// ---------------------------------------------------------------------------
// mod = silu(t_emb) @ ada_w + ada_b    (4 x 1024 @ 1024 x 6144, fp32)
// ---------------------------------------------------------------------------
__global__ __launch_bounds__(256)
void mod_gemm_kernel(const float* __restrict__ t_emb, const float* __restrict__ ada_w,
                     const float* __restrict__ ada_b, float* __restrict__ mod) {
    int b = blockIdx.y;
    int col = blockIdx.x * 256 + threadIdx.x;      // < 6144
    __shared__ float st[1024];
    for (int i = threadIdx.x; i < 1024; i += 256) {
        float v = t_emb[b * 1024 + i];
        st[i] = v / (1.f + expf(-v));              // silu
    }
    __syncthreads();
    float acc = ada_b[col];
    for (int k = 0; k < 1024; ++k) acc += st[k] * ada_w[(size_t)k * 6144 + col];
    mod[(size_t)b * 6144 + col] = acc;
}

// ---------------------------------------------------------------------------
// LayerNorm (+ optional adaLN scale/shift), fp32 in -> bf16 out. D = 1024.
// ---------------------------------------------------------------------------
__global__ __launch_bounds__(256)
void ln_mod_kernel(const float* __restrict__ x, const float* __restrict__ scale,
                   const float* __restrict__ shift, unsigned short* __restrict__ out,
                   int rows_per_b) {
    const int Dd = 1024;
    int row = blockIdx.x;
    const float* xr = x + (size_t)row * Dd;
    float s = 0.f, s2 = 0.f;
    for (int i = threadIdx.x; i < Dd; i += 256) { float v = xr[i]; s += v; s2 += v * v; }
    #pragma unroll
    for (int o = 16; o >= 1; o >>= 1) { s += __shfl_xor(s, o, 32); s2 += __shfl_xor(s2, o, 32); }
    __shared__ float rs[8], rs2[8];
    int wave = threadIdx.x >> 5;
    if ((threadIdx.x & 31) == 0) { rs[wave] = s; rs2[wave] = s2; }
    __syncthreads();
    float S = 0.f, S2 = 0.f;
    #pragma unroll
    for (int w = 0; w < 8; ++w) { S += rs[w]; S2 += rs2[w]; }
    float mean = S * (1.f / 1024.f);
    float var  = S2 * (1.f / 1024.f) - mean * mean;
    float inv  = rsqrtf(var + 1e-6f);
    int b = row / rows_per_b;
    unsigned short* orow = out + (size_t)row * Dd;
    for (int i = threadIdx.x; i < Dd; i += 256) {
        float v = (xr[i] - mean) * inv;
        if (scale) v = v * (1.f + scale[(size_t)b * 6144 + i]) + shift[(size_t)b * 6144 + i];
        orow[i] = f2bf(v);
    }
}

// ---------------------------------------------------------------------------
// Tiled bf16 WMMA GEMM: C[M,N] = A[M,K](bf16,rm) @ W[K,N](bf16,rm) + bias
// Block tile 64x64, 8 waves (each wave: 16 rows x 32 cols), K-step 32.
// A tile staged via async DMA to LDS; W staged transposed.
// EPI: 0 = store bf16; 1 = exact GELU -> bf16; 2 = fp32 out = resid + gate*(acc+bias)
// ---------------------------------------------------------------------------
template<int EPI>
__global__ __launch_bounds__(256)
void gemm_bf16_kernel(const unsigned short* __restrict__ A,
                      const unsigned short* __restrict__ W,
                      const float* __restrict__ bias,
                      int M, int K, int Nw,
                      unsigned short* __restrict__ outh,
                      float* __restrict__ outf,
                      const float* __restrict__ resid,
                      const float* __restrict__ gate,
                      int rows_per_b) {
    __shared__ alignas(16) unsigned short sA[64 * 40];   // A tile, row-major, stride 40
    __shared__ alignas(16) unsigned short sB[64 * 40];   // W tile transposed: sB[n][k]
    int tM = blockIdx.x * 64;
    int tN = blockIdx.y * 64;
    int tid = threadIdx.x;
    int wave = tid >> 5, lane = tid & 31;
    int l16 = lane & 15;
    int half16 = (lane < 16) ? 0 : 16;
    int half8  = (lane < 16) ? 0 : 8;
    int wr = (wave & 3) * 16;       // wave row offset in tile
    int wc = (wave >> 2) * 32;      // wave col offset (0 or 32)
    v8f acc0 = {}, acc1 = {};
    int am = tid >> 2;              // 0..63 rows of A tile
    int ak = (tid & 3) * 8;         // 0,8,16,24
    int wk = tid >> 3;              // 0..31 rows of W tile
    int wn = (tid & 7) * 8;         // 0..56
    int gm = tM + am;

    for (int k0 = 0; k0 < K; k0 += 32) {
        // stage A (64x32) -- async DMA to LDS
#if USE_ASYNC_LDS
        if (gm < M) {
            async_g2l_b128(A + (size_t)gm * K + k0 + ak, &sA[am * 40 + ak]);
        } else {
            *(uint4*)(&sA[am * 40 + ak]) = make_uint4(0, 0, 0, 0);
        }
#else
        {
            uint4 av = make_uint4(0, 0, 0, 0);
            if (gm < M) av = *(const uint4*)(A + (size_t)gm * K + k0 + ak);
            *(uint4*)(&sA[am * 40 + ak]) = av;
        }
#endif
        // stage W (32x64) transposed into sB[n][k]
        {
            uint4 wv = *(const uint4*)(W + (size_t)(k0 + wk) * Nw + tN + wn);
            alignas(16) unsigned short tmp[8];
            *(uint4*)tmp = wv;
            #pragma unroll
            for (int j = 0; j < 8; ++j) sB[(wn + j) * 40 + wk] = tmp[j];
        }
        // prefetch next K-step (global_prefetch_b8)
        if (k0 + 32 < K) {
            if (gm < M) __builtin_prefetch(A + (size_t)gm * K + k0 + 32 + ak, 0, 1);
            __builtin_prefetch(W + (size_t)(k0 + 32 + wk) * Nw + tN + wn, 0, 1);
        }
#if USE_ASYNC_LDS
        wait_async0();
#endif
        __syncthreads();

        BF16Frag a, b0, b1;
        const unsigned short* ap = &sA[(wr + l16) * 40 + half8];
        a.q[0] = *(const uint4*)(ap);
        a.q[1] = *(const uint4*)(ap + 16);
        const unsigned short* bp0 = &sB[(wc + l16) * 40 + half8];
        b0.q[0] = *(const uint4*)(bp0);
        b0.q[1] = *(const uint4*)(bp0 + 16);
        const unsigned short* bp1 = &sB[(wc + 16 + l16) * 40 + half8];
        b1.q[0] = *(const uint4*)(bp1);
        b1.q[1] = *(const uint4*)(bp1 + 16);

        acc0 = wmma_bf16(a, b0, acc0);
        acc1 = wmma_bf16(a, b1, acc1);
        __syncthreads();
    }

    #pragma unroll
    for (int t = 0; t < 2; ++t) {
        v8f acc = (t == 0) ? acc0 : acc1;
        int col = tN + wc + t * 16 + l16;
        float bv = bias[col];
        #pragma unroll
        for (int r = 0; r < 8; ++r) {
            int gm2 = tM + wr + r + half8;
            if (gm2 >= M) continue;
            float v = acc[r] + bv;
            if (EPI == 0) {
                outh[(size_t)gm2 * Nw + col] = f2bf(v);
            } else if (EPI == 1) {
                float g = 0.5f * v * (1.f + erff(v * 0.70710678118f));
                outh[(size_t)gm2 * Nw + col] = f2bf(g);
            } else {
                float gv = 1.f;
                if (gate) gv = gate[(size_t)(gm2 / rows_per_b) * 6144 + col];
                outf[(size_t)gm2 * Nw + col] = resid[(size_t)gm2 * Nw + col] + gv * v;
            }
        }
    }
}

// ---------------------------------------------------------------------------
// Flash attention (self & cross). Block = 4 waves, each wave owns 16 queries.
// Key loop in blocks of 32. K staged row-major (async DMA),
// V staged TRANSPOSED (sVt[hd][key]) so the PV B-fragment is 2 x ds_load_b128.
// ---------------------------------------------------------------------------
__global__ __launch_bounds__(128)
void attn_kernel(const unsigned short* __restrict__ qb, int q_stride,
                 const unsigned short* __restrict__ kvb, int kv_stride,
                 int k_col0, int v_col0,
                 const unsigned char* __restrict__ mask,
                 int nkeys, int q_rows,
                 unsigned short* __restrict__ out, int out_stride) {
    const float SCL = 0.125f;                    // 64^-0.5
    int b = blockIdx.z, h = blockIdx.y;
    int wave = threadIdx.x >> 5;
    int lane = threadIdx.x & 31;
    int l16 = lane & 15;
    int half16 = (lane < 16) ? 0 : 16;
    int half8  = (lane < 16) ? 0 : 8;
    __shared__ alignas(16) unsigned short sK[32 * 72];   // sK[key][hd]
    __shared__ alignas(16) unsigned short sVt[64 * 40];  // sVt[hd][key]
    __shared__ alignas(16) unsigned short sP[4][16 * 40];
    int q0 = blockIdx.x * 64 + wave * 16;

    // Q fragments (hd 0..31 and 32..63), kept in registers
    BF16Frag qf0, qf1;
    {
        const unsigned short* qr =
            qb + (size_t)(b * q_rows + q0 + l16) * q_stride + h * 64;
        qf0.q[0] = *(const uint4*)(qr + half8);
        qf0.q[1] = *(const uint4*)(qr + half8 + 16);
        qf1.q[0] = *(const uint4*)(qr + 32 + half8);
        qf1.q[1] = *(const uint4*)(qr + 32 + half8 + 16);
    }

    float mrow[8], lrow[8];
    v8f zero = {};
    v8f oacc[4] = {zero, zero, zero, zero};
    #pragma unroll
    for (int r = 0; r < 8; ++r) { mrow[r] = -INFINITY; lrow[r] = 0.f; }

    int nkb = (nkeys + 31) / 32;
    for (int kblk = 0; kblk < nkb; ++kblk) {
        int ks = kblk * 32;
        __syncthreads();
        // stage K (row-major) and V (transposed) tiles: 32 keys x 64 hd, bf16
        #pragma unroll
        for (int rep = 0; rep < 2; ++rep) {
            int idx = threadIdx.x + rep * 128;
            int key = idx >> 3;
            int c = (idx & 7) * 8;
            int gk = ks + key;
            if (gk < nkeys) {
                const unsigned short* kr =
                    kvb + (size_t)(b * nkeys + gk) * kv_stride + h * 64;
#if USE_ASYNC_LDS
                async_g2l_b128(kr + k_col0 + c, &sK[key * 72 + c]);
#else
                *(uint4*)(&sK[key * 72 + c]) = *(const uint4*)(kr + k_col0 + c);
#endif
                alignas(16) unsigned short tv[8];
                *(uint4*)tv = *(const uint4*)(kr + v_col0 + c);
                #pragma unroll
                for (int j = 0; j < 8; ++j) sVt[(c + j) * 40 + key] = tv[j];
            } else {
                *(uint4*)(&sK[key * 72 + c]) = make_uint4(0, 0, 0, 0);
                #pragma unroll
                for (int j = 0; j < 8; ++j) sVt[(c + j) * 40 + key] = 0;
            }
        }
#if USE_ASYNC_LDS
        wait_async0();
#endif
        __syncthreads();

        // S = Q @ K^T for two 16-key tiles (K-dim = hd, contiguous in sK rows)
        v8f s0 = {}, s1 = {};
        {
            BF16Frag bk;
            const unsigned short* k0p = &sK[l16 * 72 + half8];
            bk.q[0] = *(const uint4*)(k0p);
            bk.q[1] = *(const uint4*)(k0p + 16);
            s0 = wmma_bf16(qf0, bk, s0);
            bk.q[0] = *(const uint4*)(k0p + 32);
            bk.q[1] = *(const uint4*)(k0p + 48);
            s0 = wmma_bf16(qf1, bk, s0);
            const unsigned short* k1p = &sK[(16 + l16) * 72 + half8];
            bk.q[0] = *(const uint4*)(k1p);
            bk.q[1] = *(const uint4*)(k1p + 16);
            s1 = wmma_bf16(qf0, bk, s1);
            bk.q[0] = *(const uint4*)(k1p + 32);
            bk.q[1] = *(const uint4*)(k1p + 48);
            s1 = wmma_bf16(qf1, bk, s1);
        }

        int key0 = ks + l16;
        int key1 = ks + 16 + l16;
        bool m0 = (key0 >= nkeys) || (mask && mask[(size_t)b * nkeys + key0]);
        bool m1 = (key1 >= nkeys) || (mask && mask[(size_t)b * nkeys + key1]);
        float a0[8], a1[8], alpha[8];
        #pragma unroll
        for (int r = 0; r < 8; ++r) {
            a0[r] = m0 ? -INFINITY : s0[r] * SCL;
            a1[r] = m1 ? -INFINITY : s1[r] * SCL;
        }
        #pragma unroll
        for (int r = 0; r < 8; ++r) {
            float v = fmaxf(a0[r], a1[r]);
            v = fmaxf(v, __shfl_xor(v, 1, 32));
            v = fmaxf(v, __shfl_xor(v, 2, 32));
            v = fmaxf(v, __shfl_xor(v, 4, 32));
            v = fmaxf(v, __shfl_xor(v, 8, 32));
            float mnew = fmaxf(mrow[r], v);
            alpha[r] = (mrow[r] == -INFINITY) ? 0.f : expf(mrow[r] - mnew);
            mrow[r] = mnew;
        }
        #pragma unroll
        for (int r = 0; r < 8; ++r) {
            float p0 = (a0[r] == -INFINITY) ? 0.f : expf(a0[r] - mrow[r]);
            float p1 = (a1[r] == -INFINITY) ? 0.f : expf(a1[r] - mrow[r]);
            float ps = p0 + p1;
            ps += __shfl_xor(ps, 1, 32);
            ps += __shfl_xor(ps, 2, 32);
            ps += __shfl_xor(ps, 4, 32);
            ps += __shfl_xor(ps, 8, 32);
            lrow[r] = lrow[r] * alpha[r] + ps;
            #pragma unroll
            for (int t = 0; t < 4; ++t) oacc[t][r] = oacc[t][r] * alpha[r];
            int rr = r + half8;
            sP[wave][rr * 40 + l16]      = f2bf(p0);
            sP[wave][rr * 40 + 16 + l16] = f2bf(p1);
        }
        __syncthreads();

        // O += P(16x32) @ V(32x64): A-frag from sP, B-frags contiguous from sVt
        BF16Frag pf;
        {
            const unsigned short* pp = &sP[wave][l16 * 40 + half8];
            pf.q[0] = *(const uint4*)(pp);
            pf.q[1] = *(const uint4*)(pp + 16);
        }
        #pragma unroll
        for (int t = 0; t < 4; ++t) {
            BF16Frag vf;
            const unsigned short* vp = &sVt[(t * 16 + l16) * 40 + half8];
            vf.q[0] = *(const uint4*)(vp);
            vf.q[1] = *(const uint4*)(vp + 16);
            oacc[t] = wmma_bf16(pf, vf, oacc[t]);
        }
    }

    float linv[8];
    #pragma unroll
    for (int r = 0; r < 8; ++r) linv[r] = 1.f / lrow[r];
    #pragma unroll
    for (int t = 0; t < 4; ++t) {
        #pragma unroll
        for (int r = 0; r < 8; ++r) {
            int grow = q0 + r + half8;
            out[(size_t)(b * q_rows + grow) * out_stride + h * 64 + t * 16 + l16] =
                f2bf(oacc[t][r] * linv[r]);
        }
    }
}

// ---------------------------------------------------------------------------
// Host orchestration
// ---------------------------------------------------------------------------
extern "C" void kernel_launch(void* const* d_in, const int* in_sizes, int n_in,
                              void* d_out, int out_size, void* d_ws, size_t ws_size,
                              hipStream_t stream) {
    (void)in_sizes; (void)n_in; (void)out_size; (void)ws_size;
    const float* x      = (const float*)d_in[0];
    const float* t_emb  = (const float*)d_in[1];
    const float* ctx    = (const float*)d_in[2];
    const unsigned char* cmask = (const unsigned char*)d_in[3];
    const float* ada_w  = (const float*)d_in[4];
    const float* ada_b  = (const float*)d_in[5];
    const float* qkv_w  = (const float*)d_in[6];
    const float* qkv_b  = (const float*)d_in[7];
    const float* saw    = (const float*)d_in[8];
    const float* sab    = (const float*)d_in[9];
    const float* qw     = (const float*)d_in[10];
    const float* qb_    = (const float*)d_in[11];
    const float* kvw    = (const float*)d_in[12];
    const float* kvb    = (const float*)d_in[13];
    const float* caw    = (const float*)d_in[14];
    const float* cab    = (const float*)d_in[15];
    const float* f1w    = (const float*)d_in[16];
    const float* f1b    = (const float*)d_in[17];
    const float* f2w    = (const float*)d_in[18];
    const float* f2b    = (const float*)d_in[19];
    float* outp = (float*)d_out;

    // ws layout (bf16 weights, then scratch; regions reused across phases)
    unsigned short* w_qkv = (unsigned short*)d_ws;
    unsigned short* w_sa  = w_qkv + (size_t)1024 * 3072;
    unsigned short* w_q   = w_sa  + (size_t)1024 * 1024;
    unsigned short* w_kv  = w_q   + (size_t)1024 * 1024;
    unsigned short* w_ca  = w_kv  + (size_t)1024 * 2048;
    unsigned short* w_f1  = w_ca  + (size_t)1024 * 1024;
    unsigned short* w_f2  = w_f1  + (size_t)1024 * 4096;
    unsigned short* ctx16 = w_f2  + (size_t)4096 * 1024;          // 4*77*1024
    float* modv           = (float*)(ctx16 + (size_t)4 * 77 * 1024);
    unsigned short* act16 = (unsigned short*)(modv + (size_t)4 * 6144);  // xm1/xn2/xm3
    unsigned short* big   = act16 + (size_t)4096 * 1024;          // 32MB multi-use
    unsigned short* qkv   = big;                                  // 4096 x 3072
    unsigned short* qc    = big;                                  // 4096 x 1024 (after qkv dead)
    unsigned short* kvc   = big + (size_t)4096 * 1024;            // 308 x 2048
    unsigned short* att_o = big + (size_t)4096 * 3072;            // 4096 x 1024 (sa, then ca)
    unsigned short* hbuf  = big;                                  // 4096 x 4096 (after attn dead)
    float* x1 = (float*)(big + (size_t)4096 * 4096);
    float* x2 = x1 + (size_t)4096 * 1024;

    // 1) convert weights + context to bf16
    f32_to_bf16_kernel<<<1024, 256, 0, stream>>>(qkv_w, w_qkv, 1024L * 3072);
    f32_to_bf16_kernel<<<1024, 256, 0, stream>>>(saw,   w_sa,  1024L * 1024);
    f32_to_bf16_kernel<<<1024, 256, 0, stream>>>(qw,    w_q,   1024L * 1024);
    f32_to_bf16_kernel<<<1024, 256, 0, stream>>>(kvw,   w_kv,  1024L * 2048);
    f32_to_bf16_kernel<<<1024, 256, 0, stream>>>(caw,   w_ca,  1024L * 1024);
    f32_to_bf16_kernel<<<1024, 256, 0, stream>>>(f1w,   w_f1,  1024L * 4096);
    f32_to_bf16_kernel<<<1024, 256, 0, stream>>>(f2w,   w_f2,  4096L * 1024);
    f32_to_bf16_kernel<<<1024, 256, 0, stream>>>(ctx,   ctx16, 4L * 77 * 1024);

    // 2) adaLN modulation vector
    mod_gemm_kernel<<<dim3(24, 4), 256, 0, stream>>>(t_emb, ada_w, ada_b, modv);

    // 3) xm1 = ln(x)*(1+scale_sa)+shift_sa -> bf16
    ln_mod_kernel<<<4096, 256, 0, stream>>>(x, modv + 1024, modv + 0, act16, 1024);

    // 4) qkv = xm1 @ Wqkv + b  (bf16 out)
    gemm_bf16_kernel<0><<<dim3(64, 48), 256, 0, stream>>>(
        act16, w_qkv, qkv_b, 4096, 1024, 3072, qkv, nullptr, nullptr, nullptr, 1024);

    // 5) self-attention -> att_o (bf16)
    attn_kernel<<<dim3(16, 16, 4), 128, 0, stream>>>(
        qkv, 3072, qkv, 3072, 1024, 2048, nullptr, 1024, 1024, att_o, 1024);

    // 6) x1 = x + gate_sa * (att_o @ Wsa + b)
    gemm_bf16_kernel<2><<<dim3(64, 16), 256, 0, stream>>>(
        att_o, w_sa, sab, 4096, 1024, 1024, nullptr, x1, x, modv + 2048, 1024);

    // 7) xn2 = ln(x1)
    ln_mod_kernel<<<4096, 256, 0, stream>>>(x1, nullptr, nullptr, act16, 1024);

    // 8) qc = xn2 @ Wq + b
    gemm_bf16_kernel<0><<<dim3(64, 16), 256, 0, stream>>>(
        act16, w_q, qb_, 4096, 1024, 1024, qc, nullptr, nullptr, nullptr, 1024);

    // 9) kvc = context @ Wkv + b   (M = 308)
    gemm_bf16_kernel<0><<<dim3(5, 32), 256, 0, stream>>>(
        ctx16, w_kv, kvb, 308, 1024, 2048, kvc, nullptr, nullptr, nullptr, 1024);

    // 10) cross-attention (S = 77, masked) -> att_o
    attn_kernel<<<dim3(16, 16, 4), 128, 0, stream>>>(
        qc, 1024, kvc, 2048, 0, 1024, cmask, 77, 1024, att_o, 1024);

    // 11) x2 = x1 + (att_o @ Wca + b)
    gemm_bf16_kernel<2><<<dim3(64, 16), 256, 0, stream>>>(
        att_o, w_ca, cab, 4096, 1024, 1024, nullptr, x2, x1, nullptr, 1024);

    // 12) xm3 = ln(x2)*(1+scale_ff)+shift_ff
    ln_mod_kernel<<<4096, 256, 0, stream>>>(x2, modv + 4096, modv + 3072, act16, 1024);

    // 13) h = gelu(xm3 @ Wfc1 + b) -> bf16
    gemm_bf16_kernel<1><<<dim3(64, 64), 256, 0, stream>>>(
        act16, w_f1, f1b, 4096, 1024, 4096, hbuf, nullptr, nullptr, nullptr, 1024);

    // 14) out = x2 + gate_ff * (h @ Wfc2 + b)
    gemm_bf16_kernel<2><<<dim3(64, 16), 256, 0, stream>>>(
        hbuf, w_f2, f2b, 4096, 4096, 1024, nullptr, outp, x2, modv + 5120, 1024);
}
